// ParametricInhibition_22359599743473
// MI455X (gfx1250) — matrix operand order repaired and checked by reading, not compiled, not run on hardware
//
#include <hip/hip_runtime.h>

// ---------------------------------------------------------------------------
// ParametricInhibition on MI455X (gfx1250, wave32)
//
//  out[b,i,h,w] = sum_j inv(I - T)[i,j] * act[b,j,h,w]
//  T = circulant(ricker(63, width, damp)), size 256.
//
//  Phase 1: spectral inversion of the circulant (O(n^2), one block), then
//           pre-pack W into the exact V_WMMA_F32_16X16X32_F16 A-operand
//           per-lane layout as f16 hi/lo split fragments (workspace, 256 KB).
//  Phase 2: memory-bound GEMM (M=256,K=256,N=200704) with f16 WMMA, 3-term
//           split product (hi*hi + lo*hi + hi*lo) for fp32-class accuracy.
//           A-fragments double-buffered through LDS with CDNA5 async
//           global->LDS loads (ASYNCcnt), X stream prefetched.
// ---------------------------------------------------------------------------

#define C_CH 256
#define PIX  3136           // 56*56
#define NB   64
#define TILES_PER_BATCH (PIX / 16)     // 196
#define NWTILES (NB * TILES_PER_BATCH) // 12544
#define NBLOCKS (NWTILES / 8)          // 1568

typedef __attribute__((ext_vector_type(16))) _Float16 v16h;
typedef __attribute__((ext_vector_type(8)))  float    v8f;

static __device__ __forceinline__ unsigned pack2h(_Float16 a, _Float16 b) {
  unsigned short ua, ub;
  __builtin_memcpy(&ua, &a, 2);
  __builtin_memcpy(&ub, &b, 2);
  return (unsigned)ua | ((unsigned)ub << 16);
}

// ---------------------------------------------------------------------------
// Phase 1: build g = first column of (I - T)^-1 via the circulant spectrum,
// then emit WMMA A-fragments.
//   ws layout (dwords): idx = sel*32768 + kt*4096 + mt*256 + lane*8 + j
//   sel: 0=hi, 1=lo ; kt: K-chunk of 32 ; mt: M-tile of 16 ; j: dword in v16h
// ---------------------------------------------------------------------------
__global__ __launch_bounds__(256)
void pi_prep_kernel(const float* __restrict__ damp_p,
                    const float* __restrict__ width_p,
                    unsigned int* __restrict__ frag) {
  __shared__ float row0[C_CH];
  __shared__ float lam[C_CH];
  __shared__ float g[C_CH];

  const int t = threadIdx.x;
  const float damp  = damp_p[0];
  const float width = width_p[0];
  const float w2 = width * width;

  // circulant first-column generator of T (center tap zeroed)
  {
    float v = 0.0f;
    int fidx = -1;
    if (t <= 31)        fidx = 31 + t;    // center + right tail
    else if (t >= 225)  fidx = t - 225;   // left tail wraps
    if (fidx >= 0) {
      float x = (float)(fidx - 31);
      float r = damp * (1.0f - x * x / w2) * __expf(-x * x / (2.0f * w2));
      if (fidx == 31) r = 0.0f;           // SELF_CONNECT = False
      v = r;
    }
    row0[t] = v;
  }
  __syncthreads();

  // eigenvalues of (I - T): lam_k = 1 - sum_m row0[m] cos(2*pi*k*m/256)
  {
    float acc = 0.0f;
    const float k = (float)t;
    for (int m = 0; m < C_CH; ++m) {
      float ph = 0.0245436926061702596754f * k * (float)m; // 2*pi/256
      acc += row0[m] * __cosf(ph);
    }
    lam[t] = 1.0f - acc;
  }
  __syncthreads();

  // inverse generator: g[n] = (1/256) sum_k cos(2*pi*n*k/256) / lam_k
  {
    float acc = 0.0f;
    const float n = (float)t;
    for (int k = 0; k < C_CH; ++k) {
      float ph = 0.0245436926061702596754f * n * (float)k;
      acc += __cosf(ph) / lam[k];
    }
    g[t] = acc * (1.0f / 256.0f);
  }
  __syncthreads();

  // W[i][c] = g[(i - c) & 255]; emit as 16-bit A-matrix 16x32 fragments.
  // ISA layout (16-bit A 16x32): lane<16 row=lane, V0..3: K={0..7}, V4..7: K={16..23}
  //                              lane>=16 row=lane-16, K offset +8 per group.
  for (int idx = t; idx < 65536; idx += 256) {
    const int j    = idx & 7;
    const int lane = (idx >> 3) & 31;
    const int mt   = (idx >> 8) & 15;
    const int kt   = (idx >> 12) & 7;
    const int sel  = (idx >> 15) & 1;

    const int i  = mt * 16 + (lane & 15);
    const int kb = ((j >= 4) ? 16 : 0) + ((lane >> 4) * 8) + (j & 3) * 2;
    const int c0 = kt * 32 + kb;

    const float w0 = g[(i - c0) & 255];
    const float w1 = g[(i - c0 - 1) & 255];

    _Float16 h0, h1;
    if (sel == 0) {
      h0 = (_Float16)w0;
      h1 = (_Float16)w1;
    } else {
      h0 = (_Float16)(w0 - (float)(_Float16)w0);
      h1 = (_Float16)(w1 - (float)(_Float16)w1);
    }
    frag[idx] = pack2h(h0, h1);
  }
}

// ---------------------------------------------------------------------------
// Phase 2: GEMM. One wave = one (batch, 16-pixel) tile, all 256 out channels.
// Workgroup double-buffers the per-K-chunk A-fragments through LDS using
// CDNA5 async global->LDS b128 loads (no VGPR round-trip, ASYNCcnt tracked).
// ---------------------------------------------------------------------------
__global__ __launch_bounds__(256)
void pi_gemm_kernel(const float* __restrict__ X,
                    float* __restrict__ Y,
                    const unsigned int* __restrict__ frag) {
  // [db][sel][mt][lane][j]  -> 2*2*16*32*8 dwords = 64 KB
  __shared__ unsigned int ldsA[2][2][16][32][8];

  const int t    = threadIdx.x;
  const int lane = t & 31;
  const int wave = t >> 5;

  const int wtile = blockIdx.x * 8 + wave;   // 0 .. 12543 (exact)
  const int b  = wtile / TILES_PER_BATCH;
  const int pt = wtile - b * TILES_PER_BATCH;
  const int p0 = pt * 16;

  const float* __restrict__ Xw = X + (size_t)b * C_CH * PIX + p0 + (lane & 15);
  float*       __restrict__ Yw = Y + (size_t)b * C_CH * PIX + p0 + (lane & 15);
  const int chalf = (lane >> 4) * 16;        // B operand: K half per lane group

  // LDS byte address of staging buffer (addrspace(3) offset = low 32 bits
  // of the flat address -> same frame the ds_load_b128 reads use).
  const unsigned lds_base =
      (unsigned)(unsigned long long)(const void*)&ldsA[0][0][0][0][0];
  const unsigned long long gbase = (unsigned long long)frag;

  v8f acc[16];
#pragma unroll
  for (int m = 0; m < 16; ++m) acc[m] = (v8f)(0.0f);

  // ---- async-stage kt fragments (32 KB = 256 thr * 8 * b128) into buf ----
  // per chunk q (sel,r): lds = base + buf*32K + sel*16K + r*4K + t*16
  //                      mem = frag  + sel*128K + kt*16K + r*4K + t*16
#define PI_STAGE_ASYNC(ktv, bufv)                                            \
  {                                                                          \
    const unsigned ldsq = lds_base + (unsigned)(bufv) * 32768u + t * 16u;    \
    const unsigned memq = (unsigned)(ktv) * 16384u + t * 16u;                \
    _Pragma("unroll")                                                        \
    for (int q = 0; q < 8; ++q) {                                            \
      const unsigned vd = ldsq + (q & 4) * 4096u + (q & 3) * 4096u;          \
      const unsigned vg = memq + (q & 4) * 32768u + (q & 3) * 4096u;         \
      asm volatile("global_load_async_to_lds_b128 %0, %1, %2"                \
                   :: "v"(vd), "v"(vg), "s"(gbase) : "memory");              \
    }                                                                        \
  }

  PI_STAGE_ASYNC(0, 0);
  asm volatile("s_wait_asynccnt 0" ::: "memory");
  __syncthreads();

#pragma unroll
  for (int kt = 0; kt < 8; ++kt) {
    const int db = kt & 1;

    // kick off async staging of the next K-chunk into the other buffer
    if (kt < 7) PI_STAGE_ASYNC(kt + 1, db ^ 1);

    // ---- build B fragments (32x16, f16): load f32, split hi/lo ----
    // lanes 0-15: K = k0+{0..7,16..23} (col = lane), lanes 16-31: K = k0+{8..15,24..31}
    const int k0 = kt * 32;
    v16h bhi, blo;
#pragma unroll
    for (int j = 0; j < 8; ++j) {
      const int c = k0 + (chalf >> 1) + ((j & 4) << 2) + 2 * (j & 3);
      const float x0 = Xw[(size_t)c * PIX];
      const float x1 = Xw[(size_t)(c + 1) * PIX];
      if (kt < 7) {
        __builtin_prefetch(&Xw[(size_t)(c + 32) * PIX], 0, 1);
        __builtin_prefetch(&Xw[(size_t)(c + 33) * PIX], 0, 1);
      }
      const _Float16 h0 = (_Float16)x0;
      const _Float16 h1 = (_Float16)x1;
      bhi[2 * j]     = h0;
      bhi[2 * j + 1] = h1;
      blo[2 * j]     = (_Float16)(x0 - (float)h0);
      blo[2 * j + 1] = (_Float16)(x1 - (float)h1);
    }

    // ---- 16 M-tiles x (hi*hi + lo*hi + hi*lo) ----
#pragma unroll
    for (int mt = 0; mt < 16; ++mt) {
      const v16h ahi = *(const v16h*)&ldsA[db][0][mt][lane][0];
      const v16h alo = *(const v16h*)&ldsA[db][1][mt][lane][0];
      acc[mt] = __builtin_amdgcn_wmma_f32_16x16x32_f16(
          false, ahi, false, bhi, (short)0, acc[mt], false, false);
      acc[mt] = __builtin_amdgcn_wmma_f32_16x16x32_f16(
          false, alo, false, bhi, (short)0, acc[mt], false, false);
      acc[mt] = __builtin_amdgcn_wmma_f32_16x16x32_f16(
          false, ahi, false, blo, (short)0, acc[mt], false, false);
    }

    // drain this iteration's async stores into LDS, then swap buffers
    asm volatile("s_wait_asynccnt 0" ::: "memory");
    __syncthreads();
  }

  // ---- epilogue: D layout -> channel-major f32 stores (64B row segments) --
  // VGPR r: lanes 0-15 -> M = r, lanes 16-31 -> M = r+8; N = lane & 15.
#pragma unroll
  for (int mt = 0; mt < 16; ++mt) {
#pragma unroll
    for (int r = 0; r < 8; ++r) {
      const int m = mt * 16 + r + (lane >> 4) * 8;
      Yw[(size_t)m * PIX] = acc[mt][r];
    }
  }
}

// ---------------------------------------------------------------------------
extern "C" void kernel_launch(void* const* d_in, const int* in_sizes, int n_in,
                              void* d_out, int out_size, void* d_ws, size_t ws_size,
                              hipStream_t stream) {
  const float* act   = (const float*)d_in[0];
  const float* damp  = (const float*)d_in[1];
  const float* width = (const float*)d_in[2];
  unsigned int* frag = (unsigned int*)d_ws;   // needs 256 KiB scratch

  pi_prep_kernel<<<1, 256, 0, stream>>>(damp, width, frag);
  pi_gemm_kernel<<<NBLOCKS, 256, 0, stream>>>(act, (float*)d_out, frag);
}
